// HadamardHeadMixer_54073638256921
// MI455X (gfx1250) — matrix operand order repaired and checked by reading, not compiled, not run on hardware
//
#include <hip/hip_runtime.h>

// ---------------------------------------------------------------------------
// HadamardHeadMixer for MI455X (gfx1250, wave32, WMMA bf16 16x16x32)
//   out[b,g,t,o] = beta[o]/32 * sum_h Hu[g,h] * ( (sum_h' Hu[h,h'] x[b,h',t,:]) @ W[h] )[o]
// Hu = unnormalized +/-1 Sylvester Hadamard; Hu[a][b] = (-1)^popc(a&b).
// ---------------------------------------------------------------------------

typedef __attribute__((ext_vector_type(16))) __bf16       v16bf;
typedef __attribute__((ext_vector_type(2)))  __bf16       bf16x2;
typedef __attribute__((ext_vector_type(8)))  float        v8f;
typedef __attribute__((ext_vector_type(4)))  unsigned int u32x4;

#define TN 4096   // T
#define DD 128    // head dim
#define NH 32     // algebra dim (heads)
#define TT 16     // t-rows per workgroup tile

union FragBF { u32x4 q[2]; v16bf v; };        // trivially-constructible union
union FragHU { unsigned int u[8]; v16bf v; };

static __device__ __forceinline__ v8f wmma_bf16(v16bf a, v16bf b, v8f c) {
  // D = A(16x32 bf16) x B(32x16 bf16) + C(16x16 f32)
  return __builtin_amdgcn_wmma_f32_16x16x32_bf16(
      /*neg_a=*/false, a, /*neg_b=*/false, b,
      /*c_mod=*/(short)0, c, /*reuse_a=*/false, /*reuse_b=*/false);
}

// Load a 32-byte fragment half-pair (two b128 loads) from LDS or global.
static __device__ __forceinline__ v16bf ld_frag32(const char* p) {
  FragBF f;
  f.q[0] = *(const u32x4*)(p);
  f.q[1] = *(const u32x4*)(p + 16);
  return f.v;
}

static __device__ __forceinline__ bf16x2 pk2(float a, float b) {
  bf16x2 e;
  e.x = (__bf16)a;
  e.y = (__bf16)b;
  return e;
}

// --------------------------- W pre-pass ------------------------------------
// W fp32 [h][i][o]  ->  Wt bf16 [h][o][i]   (i contiguous => B-fragment ready)
__global__ __launch_bounds__(256, 1)
void hhm_wcvt(const float* __restrict__ W, unsigned short* __restrict__ Wt) {
  int idx = blockIdx.x * 256 + threadIdx.x;   // 32*128*128 = 524288 elements
  int i = idx & 127;
  int o = (idx >> 7) & 127;
  int h = idx >> 14;
  __bf16* wt = (__bf16*)Wt;
  wt[(h << 14) + (o << 7) + i] = (__bf16)W[(h << 14) + (i << 7) + o];
}

// --------------------------- main kernel -----------------------------------
// grid = B * (TN/TT) blocks of 256 threads (8 waves). 256 KB dynamic LDS:
//   xs : bf16 [t][d][h]   (16*128*32)  -- x slab, later reused for y [t][o][h]
//   xm : bf16 [h][t][i]   (32*16*128)  -- mixed x, A-fragment layout
__global__ __launch_bounds__(256, 1)
void hhm_main(const float* __restrict__ x, const unsigned short* __restrict__ Wtg,
              const float* __restrict__ beta, float* __restrict__ out) {
  extern __shared__ char smem[];
  __bf16* xs = (__bf16*)smem;                  // 131072 B
  __bf16* xm = (__bf16*)(smem + 131072);       // 131072 B

  const int tid  = threadIdx.x;
  const int lane = tid & 31;
  const int wave = tid >> 5;
  const int b    = blockIdx.x >> 8;            // TN/TT = 256 tiles per batch
  const int t0   = (blockIdx.x & 255) << 4;
  const int ln16 = lane & 15;
  const int hiHf = (lane >= 16) ? 1 : 0;       // lane half selector

  // ---- Phase 0: global x (fp32) -> LDS xs (bf16, [t][d][h]) ----
  // Head-PAIR staging: pack (h, h+1) bf16 into one dword -> ds_store_b32.
  {
    const float* xb = x + ((size_t)(b * NH) * TN + t0) * DD;
    for (int it = 0; it < 32; ++it) {
      int c  = tid + (it << 8);                // 0..8191 pair-chunks
      int hp = c >> 9;                         // head pair 0..15
      int t  = (c >> 5) & 15;
      int dq = (c & 31) << 2;
      const float* p0 = xb + ((size_t)(hp << 1) * TN + t) * DD + dq;
      const float4 v0 = *(const float4*)(p0);
      const float4 v1 = *(const float4*)(p0 + (size_t)TN * DD);
      int base = (t * DD + dq) * NH + (hp << 1);
      *(bf16x2*)(xs + base)          = pk2(v0.x, v1.x);
      *(bf16x2*)(xs + base + NH)     = pk2(v0.y, v1.y);
      *(bf16x2*)(xs + base + 2 * NH) = pk2(v0.z, v1.z);
      *(bf16x2*)(xs + base + 3 * NH) = pk2(v0.w, v1.w);
    }
  }

  // ---- Constant Hadamard A-fragments (g rows 0..15 and 16..31) ----
  // A 16x32 fragment in this code's contiguous k order (matching B): lanes hold M = lane&15; VGPR j covers
  //   K = (lane>=16?16:0) + 2*j (+1 in high half of dword).
  FragHU hu0, hu1;
  {
    int m0 = ln16;
    int m1 = 16 + ln16;
    int ka = hiHf ? 16 : 0;
#pragma unroll
    for (int j = 0; j < 8; ++j) {
      int k = ka + (j << 1);
      unsigned lo0 = (__popc((unsigned)(m0 & k)) & 1) ? 0xBF80u : 0x3F80u;
      unsigned hi0 = (__popc((unsigned)(m0 & (k + 1))) & 1) ? 0xBF80u : 0x3F80u;
      unsigned lo1 = (__popc((unsigned)(m1 & k)) & 1) ? 0xBF80u : 0x3F80u;
      unsigned hi1 = (__popc((unsigned)(m1 & (k + 1))) & 1) ? 0xBF80u : 0x3F80u;
      hu0.u[j] = lo0 | (hi0 << 16);
      hu1.u[j] = lo1 | (hi1 << 16);
    }
  }
  __syncthreads();

  // ---- Phase 1: mix1 = Hu (32x32) applied over heads, via WMMA ----
  // Columns = flat (t,d); 128 column-tiles split across 8 waves.
  for (int s = 0; s < 16; ++s) {
    int ct = (wave << 4) + s;
    int t  = ct >> 3;
    int dt = ct & 7;
    int d  = (dt << 4) + ln16;
    const char* p = (const char*)xs + (size_t)(t * DD + d) * (NH * 2) + (hiHf ? 32 : 0);
    v16bf bf = ld_frag32(p);
    v8f a0 = {}, a1 = {};
    a0 = wmma_bf16(hu0.v, bf, a0);             // g = 0..15
    a1 = wmma_bf16(hu1.v, bf, a1);             // g = 16..31
#pragma unroll
    for (int r = 0; r < 8; ++r) {
      int g = r + (hiHf ? 8 : 0);              // C/D: VGPR r -> M = r (+8 hi half)
      xm[(g * TT + t) * DD + d]        = (__bf16)a0[r];
      xm[((g + 16) * TT + t) * DD + d] = (__bf16)a1[r];
    }
  }
  __syncthreads();

  // ---- Phase 2: per-head y[t,o] = xm[h] (16x128) @ W[h] (128x128) ----
  // Wave w owns heads 4w..4w+3, processed as 2 pairs so y stores pack to b32
  // and two independent WMMA chains overlap with W global_load_b128s.
  for (int hp = 0; hp < 2; ++hp) {
    int h0 = (wave << 2) + (hp << 1);
    int h1 = h0 + 1;
    v16bf af0[4], af1[4];
#pragma unroll
    for (int kt = 0; kt < 4; ++kt) {           // A frag: M=t (lane&15), K=i tile
      int i0 = (kt << 5) + (hiHf ? 16 : 0);
      int i1 = (kt << 5) + (hiHf ? 24 : 8);
      const char* ab0 = (const char*)xm + (size_t)(h0 * TT + ln16) * (DD * 2);
      const char* ab1 = ab0 + (size_t)TT * DD * 2;
      FragBF f0, f1;
      f0.q[0] = *(const u32x4*)(ab0 + i0 * 2);
      f0.q[1] = *(const u32x4*)(ab0 + i1 * 2);
      f1.q[0] = *(const u32x4*)(ab1 + i0 * 2);
      f1.q[1] = *(const u32x4*)(ab1 + i1 * 2);
      af0[kt] = f0.v;
      af1[kt] = f1.v;
    }
    for (int ot = 0; ot < 8; ++ot) {
      int o = (ot << 4) + ln16;
      const char* wp0 = (const char*)Wtg +
                        ((size_t)((h0 << 7) + o) * DD) * 2 + (hiHf ? 32 : 0);
      const char* wp1 = wp0 + (size_t)DD * DD * 2;   // next head's W
      v8f acc0 = {}, acc1 = {};
#pragma unroll
      for (int kt = 0; kt < 4; ++kt) {
        acc0 = wmma_bf16(af0[kt], ld_frag32(wp0 + (kt << 6)), acc0);
        acc1 = wmma_bf16(af1[kt], ld_frag32(wp1 + (kt << 6)), acc1);
      }
#pragma unroll
      for (int r = 0; r < 8; ++r) {
        int t = r + (hiHf ? 8 : 0);
        *(bf16x2*)(xs + (t * DD + o) * NH + h0) = pk2(acc0[r], acc1[r]);
      }
    }
  }
  __syncthreads();

  // ---- Phase 3: mix2 via WMMA, scale by beta[o]/32, store fp32 ----
  for (int s = 0; s < 16; ++s) {
    int ct = (wave << 4) + s;
    int t  = ct >> 3;
    int dt = ct & 7;
    int o  = (dt << 4) + ln16;
    const char* p = (const char*)xs + (size_t)(t * DD + o) * (NH * 2) + (hiHf ? 32 : 0);
    v16bf bf = ld_frag32(p);
    v8f a0 = {}, a1 = {};
    a0 = wmma_bf16(hu0.v, bf, a0);
    a1 = wmma_bf16(hu1.v, bf, a1);
    float bv = beta[o] * 0.03125f;             // two unnormalized Hadamards -> /32
    size_t row = (size_t)(t0 + t) * DD + o;
#pragma unroll
    for (int r = 0; r < 8; ++r) {
      int g = r + (hiHf ? 8 : 0);
      out[((size_t)(b * NH + g) * TN) * DD + row]      = a0[r] * bv;
      out[((size_t)(b * NH + g + 16) * TN) * DD + row] = a1[r] * bv;
    }
  }
}

// --------------------------- host launcher ---------------------------------
extern "C" void kernel_launch(void* const* d_in, const int* in_sizes, int n_in,
                              void* d_out, int out_size, void* d_ws, size_t ws_size,
                              hipStream_t stream) {
  const float* x    = (const float*)d_in[0];   // [B,32,4096,128] fp32
  const float* W    = (const float*)d_in[1];   // [32,128,128] fp32
  const float* beta = (const float*)d_in[2];   // [128] fp32
  float* out        = (float*)d_out;
  unsigned short* Wt = (unsigned short*)d_ws;  // bf16 [32][128][128] = 1 MB

  int B = in_sizes[0] / (NH * TN * DD);

  // Allow 256 KB dynamic LDS (WGP supports 320 KB). Deterministic each call.
  (void)hipFuncSetAttribute((const void*)hhm_main,
                            hipFuncAttributeMaxDynamicSharedMemorySize, 262144);

  hipLaunchKernelGGL(hhm_wcvt, dim3((NH * DD * DD) / 256), dim3(256), 0, stream, W, Wt);
  hipLaunchKernelGGL(hhm_main, dim3(B * (TN / TT)), dim3(256), 262144, stream,
                     x, Wt, beta, out);
}
